// SSCA_3796751090401
// MI455X (gfx1250) — matrix operand order repaired and checked by reading, not compiled, not run on hardware
//
#include <hip/hip_runtime.h>
#include <hip/hip_bf16.h>
#include <math.h>

typedef unsigned int u32;
typedef unsigned short u16;
typedef __bf16 bf16;
typedef bf16 v16bf __attribute__((ext_vector_type(16)));
typedef float v8f  __attribute__((ext_vector_type(8)));
typedef u32  u32x4 __attribute__((ext_vector_type(4)));

struct FragQ { u32x4 lo; u32x4 hi; };
union Frag { FragQ q; v16bf v; };

__device__ __forceinline__ u16 f2bf(float f) {
  u32 u = __builtin_bit_cast(u32, f);
  u += 0x7FFFu + ((u >> 16) & 1u);   // round-to-nearest-even
  return (u16)(u >> 16);
}
__device__ __forceinline__ float bf2f(u16 h) {
  return __builtin_bit_cast(float, (u32)h << 16);
}
__device__ __forceinline__ float ldf(float v) { return v; }
__device__ __forceinline__ float ldf(u16 v)  { return bf2f(v); }

// ---------------- elementwise f32 -> bf16 (weights, one-shot) --------------
__global__ __launch_bounds__(256)
void ssca_f32_to_bf16(const float* __restrict__ in, u16* __restrict__ out, int n) {
  int i = blockIdx.x * 256 + threadIdx.x;
  if (i < n) out[i] = f2bf(in[i]);
}

// ---------------- depthwise 3x3 + bias -> bf16 [B,C,64,64] ----------------
__global__ __launch_bounds__(256)
void ssca_dw3x3(const float* __restrict__ x, const float* __restrict__ w,
                const float* __restrict__ bias, u16* __restrict__ y, int C) {
  int n = blockIdx.x * 256 + threadIdx.x;   // 0..4095
  int c = blockIdx.y;
  int b = blockIdx.z;
  int h = n >> 6, wc = n & 63;
  const float* xp = x + (((long)b * C + c) << 12);
  float wr[9];
#pragma unroll
  for (int i = 0; i < 9; ++i) wr[i] = w[c * 9 + i];
  float acc = bias[c];
#pragma unroll
  for (int dy = -1; dy <= 1; ++dy)
#pragma unroll
    for (int dx = -1; dx <= 1; ++dx) {
      int hh = h + dy, ww = wc + dx;
      if (hh >= 0 && hh < 64 && ww >= 0 && ww < 64)
        acc += wr[(dy + 1) * 3 + (dx + 1)] * xp[(hh << 6) + ww];
    }
  y[(((long)b * C + c) << 12) + n] = f2bf(acc);
}

// ------------- LDS-tiled transpose: in[r,c] -> out[c,r] (bf16 out) --------
template <typename TIN>
__global__ __launch_bounds__(256)
void ssca_transpose_bf16(const TIN* __restrict__ in, u16* __restrict__ out,
                         int rows, int cols) {
  __shared__ float tile[32][33];
  long zoff = (long)blockIdx.z * rows * cols;
  int c0 = blockIdx.x * 32, r0 = blockIdx.y * 32;
  int tx = threadIdx.x, ty = threadIdx.y;  // block (32,8)
#pragma unroll
  for (int j = 0; j < 4; ++j) {
    int r = r0 + ty + j * 8, c = c0 + tx;
    if (r < rows && c < cols) tile[ty + j * 8][tx] = ldf(in[zoff + (long)r * cols + c]);
  }
  __syncthreads();
#pragma unroll
  for (int j = 0; j < 4; ++j) {
    int r = r0 + tx, c = c0 + ty + j * 8;
    if (r < rows && c < cols) out[zoff + (long)c * rows + r] = f2bf(tile[tx][ty + j * 8]);
  }
}

// ---------------- bf16 WMMA GEMM: C = alpha*(A@B) + bias -------------------
// A: [M,K] row-major bf16, lda = row stride (K-contiguous)
// B: column-major bf16, element B[k][j] = Bp[j*ldb + k] (K-contiguous)
// Each wave computes a 32x32 output tile (2x2 wmma tiles, 4 independent
// accumulators per K-step). Block = 4 waves along N => block tile 32M x 128N.
template <bool HASBIAS>
__global__ __launch_bounds__(128)
void ssca_gemm_wmma(const u16* __restrict__ Ap, const u16* __restrict__ Bp,
                    const float* __restrict__ bias, float* __restrict__ Cf,
                    int M, int N, int K, int lda, int ldb, int ldc,
                    long sA, long sB, long sC, float alpha) {
  int z = blockIdx.z;
  int wave = threadIdx.x >> 5, lane = threadIdx.x & 31;
  int m0 = blockIdx.y << 5;                     // 32 rows per wave (grid exact)
  int n0 = (blockIdx.x << 7) + (wave << 5);     // 32 cols per wave
  if (n0 >= N) return;                          // wave-uniform: EXEC all-ones
  int r = lane & 15, hi = lane >> 4;

  const u16* A0 = Ap + (long)z * sA + (long)(m0 + r) * lda + (hi << 3);
  const u16* A1 = A0 + (long)16 * lda;
  const u16* B0 = Bp + (long)z * sB + (long)(n0 + r) * ldb + (hi << 3);
  const u16* B1 = B0 + (long)16 * ldb;

  v8f acc[2][2];
#pragma unroll
  for (int i = 0; i < 2; ++i)
#pragma unroll
    for (int j = 0; j < 2; ++j) acc[i][j] = (v8f){0.f,0.f,0.f,0.f,0.f,0.f,0.f,0.f};

  for (int k0 = 0; k0 < K; k0 += 32) {
    Frag a0, a1, b0, b1;
    a0.q.lo = *(const u32x4*)(A0 + k0);       // lane K-half: 8 bf16 = 16B
    a0.q.hi = *(const u32x4*)(A0 + k0 + 16);
    a1.q.lo = *(const u32x4*)(A1 + k0);
    a1.q.hi = *(const u32x4*)(A1 + k0 + 16);
    b0.q.lo = *(const u32x4*)(B0 + k0);       // lane K-half: 8 bf16 = 16B
    b0.q.hi = *(const u32x4*)(B0 + k0 + 16);
    b1.q.lo = *(const u32x4*)(B1 + k0);
    b1.q.hi = *(const u32x4*)(B1 + k0 + 16);
    acc[0][0] = __builtin_amdgcn_wmma_f32_16x16x32_bf16(false, a0.v, false, b0.v,
                                                        (short)0, acc[0][0], false, false);
    acc[0][1] = __builtin_amdgcn_wmma_f32_16x16x32_bf16(false, a0.v, false, b1.v,
                                                        (short)0, acc[0][1], false, false);
    acc[1][0] = __builtin_amdgcn_wmma_f32_16x16x32_bf16(false, a1.v, false, b0.v,
                                                        (short)0, acc[1][0], false, false);
    acc[1][1] = __builtin_amdgcn_wmma_f32_16x16x32_bf16(false, a1.v, false, b1.v,
                                                        (short)0, acc[1][1], false, false);
  }

  float* Cz = Cf + (long)z * sC;
#pragma unroll
  for (int mt = 0; mt < 2; ++mt) {
#pragma unroll
    for (int i = 0; i < 8; ++i) {
      int m = m0 + (mt << 4) + i + (hi << 3);
      float bv = 0.f;
      if (HASBIAS) bv = bias[m];
      float* row = Cz + (long)m * ldc + n0 + r;
      row[0]  = acc[mt][0][i] * alpha + bv;
      row[16] = acc[mt][1][i] * alpha + bv;
    }
  }
}

// ---------------- row L2 normalize (4096-wide rows) -> bf16 ----------------
__global__ __launch_bounds__(256)
void ssca_l2norm(const float* __restrict__ in, u16* __restrict__ out, int rowlen) {
  __shared__ float sred[256];
  __shared__ float sscale;
  long base = (long)blockIdx.x * rowlen;
  int tid = threadIdx.x;
  float s = 0.f;
  for (int i = tid; i < rowlen; i += 256) { float v = in[base + i]; s += v * v; }
  sred[tid] = s; __syncthreads();
  for (int st = 128; st > 0; st >>= 1) {
    if (tid < st) sred[tid] += sred[tid + st];
    __syncthreads();
  }
  if (tid == 0) { float n = sqrtf(sred[0]); sscale = 1.0f / fmaxf(n, 1e-12f); }
  __syncthreads();
  float sc = sscale;
  for (int i = tid; i < rowlen; i += 256) out[base + i] = f2bf(in[base + i] * sc);
}

// ---------------- instance-norm stats over (c_i x 672) plane ---------------
__global__ __launch_bounds__(256)
void ssca_inorm_stats(const float* __restrict__ attn, float* __restrict__ stats,
                      long plane) {
  __shared__ float s1[256], s2[256];
  int tid = threadIdx.x;
  long base = (long)blockIdx.x * plane;
  float a = 0.f, b = 0.f;
  for (long i = tid; i < plane; i += 256) { float v = attn[base + i]; a += v; b += v * v; }
  s1[tid] = a; s2[tid] = b; __syncthreads();
  for (int st = 128; st > 0; st >>= 1) {
    if (tid < st) { s1[tid] += s1[tid + st]; s2[tid] += s2[tid + st]; }
    __syncthreads();
  }
  if (tid == 0) {
    float mu = s1[0] / (float)plane;
    float var = fmaxf(s2[0] / (float)plane - mu * mu, 0.f);
    stats[blockIdx.x * 2 + 0] = mu;
    stats[blockIdx.x * 2 + 1] = rsqrtf(var + 1e-5f);
  }
}

// ---------------- fused instnorm-apply + softmax over 672 -> bf16 ----------
__global__ __launch_bounds__(256)
void ssca_softmax(const float* __restrict__ attn, const float* __restrict__ stats,
                  u16* __restrict__ p, int rows) {
  __shared__ float sred[256];
  int c = blockIdx.x, b = blockIdx.y, tid = threadIdx.x;
  long base = ((long)b * rows + c) * 672;
  float mu = stats[b * 2], is = stats[b * 2 + 1];
  float av[3];
  float mx = -1e30f;
#pragma unroll
  for (int j = 0; j < 3; ++j) {
    int i = tid + j * 256;
    float v = (i < 672) ? (attn[base + i] - mu) * is : -1e30f;
    av[j] = v; mx = fmaxf(mx, v);
  }
  sred[tid] = mx; __syncthreads();
  for (int st = 128; st > 0; st >>= 1) {
    if (tid < st) sred[tid] = fmaxf(sred[tid], sred[tid + st]);
    __syncthreads();
  }
  mx = sred[0]; __syncthreads();
  float sum = 0.f;
#pragma unroll
  for (int j = 0; j < 3; ++j) {
    int i = tid + j * 256;
    if (i < 672) { av[j] = __expf(av[j] - mx); sum += av[j]; }
  }
  sred[tid] = sum; __syncthreads();
  for (int st = 128; st > 0; st >>= 1) {
    if (tid < st) sred[tid] += sred[tid + st];
    __syncthreads();
  }
  float inv = 1.0f / sred[0];
#pragma unroll
  for (int j = 0; j < 3; ++j) {
    int i = tid + j * 256;
    if (i < 672) p[base + i] = f2bf(av[j] * inv);
  }
}

// ===========================================================================
extern "C" void kernel_launch(void* const* d_in, const int* in_sizes, int n_in,
                              void* d_out, int out_size, void* d_ws, size_t ws_size,
                              hipStream_t stream) {
  (void)in_sizes; (void)n_in; (void)out_size; (void)ws_size;
  constexpr int Bn = 4, C1 = 96, C2 = 192, C3 = 384, KVC = 672;
  constexpr long NP = 4096;

  const float* emb1 = (const float*)d_in[0];
  const float* emb2 = (const float*)d_in[1];
  const float* emb3 = (const float*)d_in[2];
  const float* embA = (const float*)d_in[3];

  char* ws = (char*)d_ws;
  size_t cur = 0;
  auto take = [&](size_t bytes) { size_t o = cur; cur += (bytes + 255) & ~(size_t)255; return o; };

  const long dwElems = (long)Bn * (C1 + C2 + C3 + 2 * KVC) * NP; // 2016 ch
  size_t off_dw  = take((size_t)dwElems * 2);                    // bf16, reused as q f32
  size_t off_dwT = take((size_t)dwElems * 2);                    // bf16, reused as out f32 + outT
  size_t off_kf  = take((size_t)Bn * KVC * NP * 4);
  size_t off_vf  = take((size_t)Bn * KVC * NP * 4);
  size_t off_qn  = take((size_t)Bn * KVC * NP * 2);
  size_t off_kn  = take((size_t)Bn * KVC * NP * 2);
  size_t off_vT  = take((size_t)Bn * KVC * NP * 2);
  size_t off_at  = take((size_t)Bn * KVC * KVC * 4);
  size_t off_p   = take((size_t)Bn * KVC * KVC * 2);
  size_t off_st  = take(1024);
  size_t off_wb  = take((size_t)(C1*C1 + C2*C2 + C3*C3 + 2*KVC*KVC
                                 + C1*C1 + C2*C2 + C3*C3) * 2);  // bf16 weights

  u16*   dw    = (u16*)(ws + off_dw);
  u16*   dwT   = (u16*)(ws + off_dwT);
  float* qf    = (float*)(ws + off_dw);                              // alias of dw
  float* kf    = (float*)(ws + off_kf);
  float* vf    = (float*)(ws + off_vf);
  u16*   qn    = (u16*)(ws + off_qn);
  u16*   kn    = (u16*)(ws + off_kn);
  u16*   vT    = (u16*)(ws + off_vT);
  float* attn  = (float*)(ws + off_at);
  u16*   pbuf  = (u16*)(ws + off_p);
  float* stats = (float*)(ws + off_st);
  u16*   wb    = (u16*)(ws + off_wb);
  float* outf  = (float*)(ws + off_dwT);                             // alias of dwT
  u16*   outT  = (u16*)(ws + off_dwT + (size_t)Bn * KVC * NP * 4);

  const long s_q1 = 0;
  const long s_q2 = (long)Bn * C1 * NP;
  const long s_q3 = s_q2 + (long)Bn * C2 * NP;
  const long s_k  = s_q3 + (long)Bn * C3 * NP;
  const long s_v  = s_k + (long)Bn * KVC * NP;

  // --- one-shot f32->bf16 conversion of all weight matrices
  const int wsrc[8] = {6, 10, 14, 18, 22, 24, 25, 26};
  const int wsz[8]  = {C1*C1, C2*C2, C3*C3, KVC*KVC, KVC*KVC, C1*C1, C2*C2, C3*C3};
  long wOff[8]; { long o = 0; for (int i = 0; i < 8; ++i) { wOff[i] = o; o += wsz[i]; } }
  for (int i = 0; i < 8; ++i)
    ssca_f32_to_bf16<<<dim3((wsz[i] + 255) / 256), 256, 0, stream>>>(
        (const float*)d_in[wsrc[i]], wb + wOff[i], wsz[i]);

  struct Conv { int C; const float* x; int wi; long dwOff; };
  Conv convs[5] = { {C1, emb1, 4, s_q1}, {C2, emb2, 8, s_q2}, {C3, emb3, 12, s_q3},
                    {KVC, embA, 16, s_k}, {KVC, embA, 20, s_v} };

  // --- depthwise 3x3 -> bf16, then transpose to [B][N][C] for WMMA B-operand
  for (int i = 0; i < 5; ++i) {
    Conv& cv = convs[i];
    ssca_dw3x3<<<dim3(16, cv.C, Bn), 256, 0, stream>>>(
        cv.x, (const float*)d_in[cv.wi], (const float*)d_in[cv.wi + 1],
        dw + cv.dwOff, cv.C);
  }
  for (int i = 0; i < 5; ++i) {
    Conv& cv = convs[i];
    ssca_transpose_bf16<u16><<<dim3(128, cv.C / 32, Bn), dim3(32, 8), 0, stream>>>(
        dw + cv.dwOff, dwT + cv.dwOff, cv.C, (int)NP);
  }

  // --- pointwise 1x1 GEMMs (A = bf16 weights, B = dwT bf16), +bias -> f32
  float* pwOut[5] = { qf + s_q1, qf + s_q2, qf + s_q3, kf, vf };
  for (int i = 0; i < 5; ++i) {
    Conv& cv = convs[i];
    ssca_gemm_wmma<true><<<dim3(32, cv.C / 32, Bn), 128, 0, stream>>>(
        wb + wOff[i], dwT + cv.dwOff, (const float*)d_in[cv.wi + 3], pwOut[i],
        cv.C, (int)NP, cv.C, cv.C, cv.C, (int)NP,
        0L, (long)cv.C * NP, (long)cv.C * NP, 1.0f);
  }

  // --- L2 normalize rows of q (all branches contiguous) and k -> bf16
  ssca_l2norm<<<dim3(Bn * KVC), 256, 0, stream>>>(kf, kn, (int)NP);
  ssca_l2norm<<<dim3(Bn * KVC), 256, 0, stream>>>(qf, qn, (int)NP);

  // --- v f32 -> vT bf16 [B][N][KV]
  ssca_transpose_bf16<float><<<dim3(128, KVC / 32, Bn), dim3(32, 8), 0, stream>>>(
      vf, vT, KVC, (int)NP);

  const int  cs[3]   = {C1, C2, C3};
  const long qOff[3] = {s_q1, s_q2, s_q3};
  const long aOff[3] = {0, (long)Bn * C1 * KVC, (long)Bn * C1 * KVC + (long)Bn * C2 * KVC};
  const float alphaAttn = 1.0f / sqrtf((float)KVC);

  // --- attn = (qn @ kn^T)/sqrt(KV): K = 4096, B operand = kn (K-contiguous)
  for (int i = 0; i < 3; ++i) {
    ssca_gemm_wmma<false><<<dim3((KVC + 127) / 128, cs[i] / 32, Bn), 128, 0, stream>>>(
        qn + qOff[i], kn, nullptr, attn + aOff[i],
        cs[i], KVC, (int)NP, (int)NP, (int)NP, KVC,
        (long)cs[i] * NP, (long)KVC * NP, (long)cs[i] * KVC, alphaAttn);
  }

  // --- instance-norm stats per (branch, batch) plane, then fused softmax
  for (int i = 0; i < 3; ++i)
    ssca_inorm_stats<<<dim3(Bn), 256, 0, stream>>>(
        attn + aOff[i], stats + i * Bn * 2, (long)cs[i] * KVC);
  for (int i = 0; i < 3; ++i)
    ssca_softmax<<<dim3(cs[i], Bn), 256, 0, stream>>>(
        attn + aOff[i], stats + i * Bn * 2, pbuf + aOff[i], cs[i]);

  // --- out = p @ v: K = 672, B operand = vT (K-contiguous) -> f32
  for (int i = 0; i < 3; ++i) {
    ssca_gemm_wmma<false><<<dim3(32, cs[i] / 32, Bn), 128, 0, stream>>>(
        pbuf + aOff[i], vT, nullptr, outf + qOff[i],
        cs[i], (int)NP, KVC, KVC, KVC, (int)NP,
        (long)cs[i] * KVC, (long)KVC * NP, (long)cs[i] * NP, 1.0f);
  }

  // --- out f32 -> outT bf16 [B][N][c_i]
  for (int i = 0; i < 3; ++i)
    ssca_transpose_bf16<float><<<dim3(128, cs[i] / 32, Bn), dim3(32, 8), 0, stream>>>(
        outf + qOff[i], outT + qOff[i], cs[i], (int)NP);

  // --- final projections: O_i = p_w @ out_i  -> d_out (f32, concatenated)
  float* O = (float*)d_out;
  for (int i = 0; i < 3; ++i) {
    ssca_gemm_wmma<false><<<dim3(32, cs[i] / 32, Bn), 128, 0, stream>>>(
        wb + wOff[5 + i], outT + qOff[i], nullptr, O + qOff[i],
        cs[i], (int)NP, cs[i], cs[i], cs[i], (int)NP,
        0L, (long)cs[i] * NP, (long)cs[i] * NP, 1.0f);
  }
}
